// WaveletSparseBlock_40183714021505
// MI455X (gfx1250) — matrix operand order repaired and checked by reading, not compiled, not run on hardware
//
#include <hip/hip_runtime.h>
#include <hip/hip_bf16.h>
#include <math.h>

typedef __attribute__((ext_vector_type(16))) _Float16 v16h;
typedef __attribute__((ext_vector_type(8)))  float    v8f;
typedef __attribute__((ext_vector_type(4)))  unsigned u32x4;
typedef __attribute__((ext_vector_type(8)))  unsigned u32x8;

#define DIM    128
#define HEADS  8
#define HD     16
#define HALFHW 1024   // 32*32
#define FULLHW 4096   // 64*64
#define BATCH  8

// ---------------------------------------------------------------------------
// CDNA5 async global->LDS copy (16B per lane), tracked by ASYNCcnt.
// Inline asm: portable across ROCm-7.2 / amdgpu-toolchain builtin arity skew.
// Generic shared-pointer low 32 bits == LDS byte offset (ISA 10.2 aperture map).
// ---------------------------------------------------------------------------
__device__ __forceinline__ void async_copy16(const float* g, float* l) {
    unsigned loff = (unsigned)(unsigned long long)l;
    asm volatile("global_load_async_to_lds_b128 %0, %1, off"
                 :: "v"(loff), "v"(g) : "memory");
}
__device__ __forceinline__ void async_wait_all() {
    asm volatile("s_wait_asynccnt 0x0" ::: "memory");
}

// ---------------------------------------------------------------------------
// CDNA5 Tensor Data Mover: one 2-D tile (rows x 64 floats) global -> LDS.
// D# groups per CDNA5 ISA; 2-group form (VADDR2/3
// NULL => tensors up to 2D). Issued once per wave (TDM ignores EXEC); caller
// restricts to one wave per workgroup. Tracked by TENSORcnt.
// ---------------------------------------------------------------------------
__device__ __forceinline__ void tdm_load_tile_32x64(const float* gtile, float* ltile,
                                                    int S, int I) {
    unsigned long long ga = (unsigned long long)gtile;
    u32x4 g0;
    g0[0] = 1u;                                               // count=1 (valid), user mode
    g0[1] = (unsigned)(unsigned long long)ltile;              // lds_addr (bytes)
    g0[2] = (unsigned)(ga & 0xFFFFFFFFu);                     // global_addr[31:0]
    g0[3] = (unsigned)((ga >> 32) & 0x01FFFFFFu) | (2u << 30);// global_addr[56:32] | type=2
    u32x8 g1;
    g1[0] = 2u << 16;                                         // workgroup_mask=0, data_size=4B
    g1[1] = ((unsigned)S & 0xFFFFu) << 16;                    // tensor_dim0[15:0]
    g1[2] = ((unsigned)S >> 16) | (((unsigned)I & 0xFFFFu) << 16); // dim0[31:16] | dim1[15:0]
    g1[3] = ((unsigned)I >> 16) | (64u << 16);                // dim1[31:16] | tile_dim0=64
    g1[4] = 32u;                                              // tile_dim1=32, tile_dim2=0
    g1[5] = (unsigned)S;                                      // tensor_dim0_stride[31:0]
    g1[6] = 0u;                                               // stride[47:32] | dim1_stride lo
    g1[7] = 0u;
    asm volatile("tensor_load_to_lds %0, %1" :: "s"(g0), "s"(g1) : "memory");
}

// ---------------------------------------------------------------------------
// 1) DWT: x (B,128,64,64) -> ll (B,128,32,32), high (B,384,32,32) = [lh,hl,hh]
// ---------------------------------------------------------------------------
__global__ void dwt_kernel(const float* __restrict__ x,
                           float* __restrict__ ll, float* __restrict__ high) {
    int idx = blockIdx.x * 256 + threadIdx.x;           // B*128*1024 threads
    int j = idx & 31, i = (idx >> 5) & 31;
    int c = (idx >> 10) & 127, b = idx >> 17;
    const float* xb = x + ((size_t)b * DIM + c) * FULLHW;
    float x1 = xb[(2 * i) * 64 + 2 * j];        // even row, even col
    float x2 = xb[(2 * i + 1) * 64 + 2 * j];    // odd  row, even col
    float x3 = xb[(2 * i) * 64 + 2 * j + 1];    // even row, odd  col
    float x4 = xb[(2 * i + 1) * 64 + 2 * j + 1];
    int n = i * 32 + j;
    ll[((size_t)b * DIM + c) * HALFHW + n]         = 0.5f * ( x1 + x2 + x3 + x4);
    high[((size_t)b * 384 + c) * HALFHW + n]       = 0.5f * (-x1 - x2 + x3 + x4); // lh
    high[((size_t)b * 384 + 128 + c) * HALFHW + n] = 0.5f * (-x1 + x2 - x3 + x4); // hl
    high[((size_t)b * 384 + 256 + c) * HALFHW + n] = 0.5f * ( x1 - x2 - x3 + x4); // hh
}

// ---------------------------------------------------------------------------
// 2) Batched WMMA GEMM, double-buffered tile pipeline:
//    W tile 16x32 (2KB) via per-lane async b128 copies (ASYNCcnt),
//    X tile 32x64 (8KB) via one TDM descriptor load (TENSORcnt),
//    tile k+1 movement overlaps WMMA on tile k.
//    Y_b (O x S) = W (O x I) * X_b (I x S)
//    grid = (S/64, O/16, B), block = 128 (4 waves, one 16x16 N-tile each).
// ---------------------------------------------------------------------------
__device__ __forceinline__ void stage_tiles(const float* __restrict__ Wt,
                                            const float* __restrict__ Xb,
                                            float* tW, float* tX,
                                            int m0, int n0, int k0,
                                            int I, int S, int tid) {
    // W tile: 16 rows x 32 floats = 128 x 16B transfers (1 per thread)
    int r = tid >> 3, c4 = (tid & 7) * 4;
    async_copy16(&Wt[(size_t)(m0 + r) * I + k0 + c4], &tW[r * 32 + c4]);
    // X tile: single TDM 2-D tile load, wave 0 only
    if ((tid >> 5) == 0)
        tdm_load_tile_32x64(&Xb[(size_t)k0 * S + n0], tX, S, I);
    // far-ahead L2 prefetch of the K-tile after next
    if (k0 + 64 < I)
        __builtin_prefetch(&Xb[(size_t)(k0 + 64) * S + n0 + ((tid & 15) * 4)], 0, 1);
}

__global__ __launch_bounds__(128)
void gemm_wmma_kernel(const float* __restrict__ Wt,
                      const float* __restrict__ X,
                      float* __restrict__ Y,
                      int O, int I, int S) {
    __shared__ float tW[2][16 * 32];   // 2 x 2KB
    __shared__ float tX[2][32 * 64];   // 2 x 8KB
    int tid  = threadIdx.x;
    int wid  = tid >> 5;
    int lane = tid & 31;
    int row  = lane & 15;
    int half = lane >> 4;
    int m0 = blockIdx.y * 16;
    int n0 = blockIdx.x * 64;
    const float* Xb = X + (size_t)blockIdx.z * I * S;
    float*       Yb = Y + (size_t)blockIdx.z * O * S;

    int nk = I >> 5;
    stage_tiles(Wt, Xb, tW[0], tX[0], m0, n0, 0, I, S, tid);

    v8f acc = {};
    for (int kt = 0; kt < nk; kt++) {
        async_wait_all();                       // W tile kt resident (per-wave)
        __builtin_amdgcn_s_wait_tensorcnt(0);   // X tile kt resident (wave 0; NOP elsewhere)
        __syncthreads();                        // ...visible to all waves
        if (kt + 1 < nk)                        // overlap: move tile kt+1 into other buffer
            stage_tiles(Wt, Xb, tW[(kt + 1) & 1], tX[(kt + 1) & 1],
                        m0, n0, (kt + 1) * 32, I, S, tid);
        const float* cW = tW[kt & 1];
        const float* cX = tX[kt & 1];
        v16h a, b;
#pragma unroll
        for (int e = 0; e < 16; e++) {
            int ka = e + 8 * half + (e >= 8 ? 8 : 0);     // A 16x32 layout (ISA 7.12.2)
            a[e] = (_Float16)cW[row * 32 + ka];
            int kb = e + 8 * half + (e >= 8 ? 8 : 0);     // B 32x16 layout
            b[e] = (_Float16)cX[kb * 64 + wid * 16 + row];
        }
        acc = __builtin_amdgcn_wmma_f32_16x16x32_f16(false, a, false, b,
                                                     (short)0, acc, false, false);
    }
#pragma unroll
    for (int j = 0; j < 8; j++)
        Yb[(size_t)(m0 + j + 8 * half) * S + n0 + wid * 16 + row] = acc[j];
}

// ---------------------------------------------------------------------------
// 3) L2-normalize q and k rows (over hd=16) in place inside qkv buffer
// ---------------------------------------------------------------------------
__global__ void l2norm_kernel(float* __restrict__ qkv) {
    int idx = blockIdx.x * 256 + threadIdx.x;   // 8*2*8*1024 = 131072 threads
    int n = idx & 1023;
    int t = idx >> 10;
    int head = t & 7;  t >>= 3;
    int sel  = t & 1;  int b = t >> 1;          // sel: 0=q, 1=k
    float* p = qkv + ((size_t)b * 384 + sel * DIM + head * HD) * HALFHW + n;
    float ss = 0.f;
#pragma unroll
    for (int d = 0; d < HD; d++) { float v = p[(size_t)d * HALFHW]; ss += v * v; }
    float r = 1.f / fmaxf(sqrtf(ss), 1e-12f);
#pragma unroll
    for (int d = 0; d < HD; d++) p[(size_t)d * HALFHW] *= r;
}

// ---------------------------------------------------------------------------
// 4) Fused sparse attention: scores (WMMA) -> top-512 threshold -> softmax ->
//    P*V (WMMA).  grid = (64 row-blocks, 8 heads, 8 batch), block = 512.
// ---------------------------------------------------------------------------
__global__ __launch_bounds__(512)
void attn_kernel(const float* __restrict__ qkv, float* __restrict__ attn_out) {
    __shared__ float S[16 * 1024];              // 64 KB: scores -> probs -> out-reduce
    int tid  = threadIdx.x;
    int wid  = tid >> 5;
    int lane = tid & 31;
    int row  = lane & 15;
    int half = lane >> 4;
    int rb = blockIdx.x, head = blockIdx.y, b = blockIdx.z;

    const float* qq = qkv + ((size_t)b * 384 +       head * HD) * HALFHW;
    const float* qk = qkv + ((size_t)b * 384 + 128 + head * HD) * HALFHW;
    const float* qv = qkv + ((size_t)b * 384 + 256 + head * HD) * HALFHW;

    // ---- Phase 1: scores S[m][n] = clip(0.25 * q.k) ----
    v16h afrag;
#pragma unroll
    for (int e = 0; e < 16; e++) {
        int kd = e + 8 * half + (e >= 8 ? 8 : 0);
        afrag[e] = (kd < HD) ? (_Float16)qq[(size_t)kd * HALFHW + rb * 16 + row]
                             : (_Float16)0.f;
    }
    for (int ct = wid * 4; ct < wid * 4 + 4; ct++) {
        v16h bfrag;
#pragma unroll
        for (int e = 0; e < 16; e++) {
            int kd = e + 8 * half + (e >= 8 ? 8 : 0);
            bfrag[e] = (kd < HD) ? (_Float16)qk[(size_t)kd * HALFHW + ct * 16 + row]
                                 : (_Float16)0.f;
        }
        v8f c = {};
        c = __builtin_amdgcn_wmma_f32_16x16x32_f16(false, afrag, false, bfrag,
                                                   (short)0, c, false, false);
#pragma unroll
        for (int j = 0; j < 8; j++) {
            float v = fminf(32.f, fmaxf(-32.f, c[j] * 0.25f));
            S[(j + 8 * half) * 1024 + ct * 16 + row] = v;
        }
    }
    __syncthreads();

    // ---- Phase 2: per-row top-512 threshold + masked softmax (wave = row) ----
    {
        int r = wid;
        const float* srow = &S[r * 1024];
        float lo = -33.f, hi = 33.f;
        for (int it = 0; it < 25; it++) {
            float mid = 0.5f * (lo + hi);
            int cnt = 0;
#pragma unroll 4
            for (int j = 0; j < 32; j++)
                cnt += (srow[lane + 32 * j] >= mid) ? 1 : 0;
            for (int off = 16; off; off >>= 1) cnt += __shfl_xor(cnt, off, 32);
            if (cnt >= 512) lo = mid; else hi = mid;
        }
        float thr = lo;
        float mx = -1e30f;
#pragma unroll 4
        for (int j = 0; j < 32; j++) {
            float v = srow[lane + 32 * j];
            if (v >= thr) mx = fmaxf(mx, v);
        }
        for (int off = 16; off; off >>= 1) mx = fmaxf(mx, __shfl_xor(mx, off, 32));
        float sum = 0.f;
#pragma unroll 4
        for (int j = 0; j < 32; j++) {
            float v = srow[lane + 32 * j];
            if (v >= thr) sum += expf(v - mx);
        }
        for (int off = 16; off; off >>= 1) sum += __shfl_xor(sum, off, 32);
        float inv = 1.f / sum;
#pragma unroll 4
        for (int j = 0; j < 32; j++) {
            int i = lane + 32 * j;
            float v = srow[i];
            S[r * 1024 + i] = (v >= thr) ? expf(v - mx) * inv : 0.f;
        }
    }
    __syncthreads();

    // ---- Phase 3: out = P * V, per-wave K-partials, ds_add_f32 combine ----
    v8f acc = {};
    for (int kt = wid * 2; kt < wid * 2 + 2; kt++) {
        v16h a, bv;
#pragma unroll
        for (int e = 0; e < 16; e++) {
            int kk = e + 8 * half + (e >= 8 ? 8 : 0);
            a[e] = (_Float16)S[row * 1024 + kt * 32 + kk];
            int kb = e + 8 * half + (e >= 8 ? 8 : 0);
            bv[e] = (_Float16)qv[(size_t)row * HALFHW + kt * 32 + kb];  // col = d = row
        }
        acc = __builtin_amdgcn_wmma_f32_16x16x32_f16(false, a, false, bv,
                                                     (short)0, acc, false, false);
    }
    __syncthreads();                   // all P reads done; reuse S as 16x16 reducer
    if (tid < 256) S[tid] = 0.f;
    __syncthreads();
#pragma unroll
    for (int j = 0; j < 8; j++)
        atomicAdd(&S[(j + 8 * half) * 16 + row], acc[j]);
    __syncthreads();
    if (tid < 256) {
        int m = tid >> 4, d = tid & 15;  // query row, head-dim
        attn_out[((size_t)b * DIM + head * HD + d) * HALFHW + rb * 16 + m] = S[m * 16 + d];
    }
}

// ---------------------------------------------------------------------------
// 5) GroupNorm (+opt. SiLU/GELU, +opt. residual w/ scalar-from-ptr scale)
//    grid = (G, B), block = 256
// ---------------------------------------------------------------------------
__global__ void gn_kernel(const float* __restrict__ in, float* __restrict__ out,
                          const float* __restrict__ gamma, const float* __restrict__ beta,
                          const float* __restrict__ res, const float* __restrict__ scale_ptr,
                          float scale_c, int C, int HW, int G, int act) {
    int grp = blockIdx.x, batch = blockIdx.y;
    int cpg = C / G;
    int n = cpg * HW;
    size_t base = (size_t)batch * C * HW + (size_t)grp * cpg * HW;
    __shared__ float ssum[256], ssq[256];
    float s = 0.f, q = 0.f;
    for (int i = threadIdx.x; i < n; i += 256) {
        float v = in[base + i]; s += v; q += v * v;
    }
    ssum[threadIdx.x] = s; ssq[threadIdx.x] = q;
    __syncthreads();
    for (int off = 128; off; off >>= 1) {
        if ((int)threadIdx.x < off) {
            ssum[threadIdx.x] += ssum[threadIdx.x + off];
            ssq[threadIdx.x]  += ssq[threadIdx.x + off];
        }
        __syncthreads();
    }
    float mean = ssum[0] / n;
    float var  = ssq[0] / n - mean * mean;
    float inv  = rsqrtf(var + 1e-6f);
    float scale = scale_ptr ? scale_ptr[0] : scale_c;
    for (int i = threadIdx.x; i < n; i += 256) {
        int c = grp * cpg + i / HW;
        float v = (in[base + i] - mean) * inv * gamma[c] + beta[c];
        if (act == 1)       v = v / (1.f + expf(-v));                      // SiLU
        else if (act == 2)  v = 0.5f * v * (1.f + erff(v * 0.70710678f));  // exact GELU
        out[base + i] = res ? (res[base + i] + scale * v) : v;
    }
}

// ---------------------------------------------------------------------------
// 6) Depthwise 3x3 SAME conv on (B,384,32,32)
// ---------------------------------------------------------------------------
__global__ void dwconv_kernel(const float* __restrict__ in, const float* __restrict__ w,
                              float* __restrict__ out) {
    int idx = blockIdx.x * 256 + threadIdx.x;   // 8*384*1024 threads
    int j = idx & 31, i = (idx >> 5) & 31;
    int t = idx >> 10;
    int c = t % 384, b = t / 384;
    const float* ib = in + ((size_t)b * 384 + c) * HALFHW;
    const float* wc = w + c * 9;
    float acc = 0.f;
#pragma unroll
    for (int ki = 0; ki < 3; ki++) {
#pragma unroll
        for (int kj = 0; kj < 3; kj++) {
            int ii = i + ki - 1, jj = j + kj - 1;
            if (ii >= 0 && ii < 32 && jj >= 0 && jj < 32)
                acc += wc[ki * 3 + kj] * ib[ii * 32 + jj];
        }
    }
    out[((size_t)b * 384 + c) * HALFHW + i * 32 + j] = acc;
}

// ---------------------------------------------------------------------------
// 7) IWT + global residual: out = x + 0.1 * iwt(ll+ll_out, lh, hl, hh)
// ---------------------------------------------------------------------------
__global__ void iwt_kernel(const float* __restrict__ x,
                           const float* __restrict__ ll, const float* __restrict__ ll_out,
                           const float* __restrict__ he, float* __restrict__ out) {
    int idx = blockIdx.x * 256 + threadIdx.x;   // B*128*1024 threads
    int j = idx & 31, i = (idx >> 5) & 31;
    int c = (idx >> 10) & 127, b = idx >> 17;
    int n = i * 32 + j;
    size_t llo = ((size_t)b * DIM + c) * HALFHW + n;
    float LL = ll[llo] + ll_out[llo];
    float LH = he[((size_t)b * 384 + c) * HALFHW + n];
    float HL = he[((size_t)b * 384 + 128 + c) * HALFHW + n];
    float HH = he[((size_t)b * 384 + 256 + c) * HALFHW + n];
    float va = LL - LH - HL + HH;   // (2i,   2j)
    float vb = LL - LH + HL - HH;   // (2i+1, 2j)
    float vc = LL + LH - HL - HH;   // (2i,   2j+1)
    float vd = LL + LH + HL + HH;   // (2i+1, 2j+1)
    size_t xb = ((size_t)b * DIM + c) * FULLHW;
    out[xb + (2 * i) * 64 + 2 * j]         = x[xb + (2 * i) * 64 + 2 * j]         + 0.1f * va;
    out[xb + (2 * i + 1) * 64 + 2 * j]     = x[xb + (2 * i + 1) * 64 + 2 * j]     + 0.1f * vb;
    out[xb + (2 * i) * 64 + 2 * j + 1]     = x[xb + (2 * i) * 64 + 2 * j + 1]     + 0.1f * vc;
    out[xb + (2 * i + 1) * 64 + 2 * j + 1] = x[xb + (2 * i + 1) * 64 + 2 * j + 1] + 0.1f * vd;
}

// ---------------------------------------------------------------------------
// Host-side orchestration
// ---------------------------------------------------------------------------
extern "C" void kernel_launch(void* const* d_in, const int* in_sizes, int n_in,
                              void* d_out, int out_size, void* d_ws, size_t ws_size,
                              hipStream_t stream) {
    (void)in_sizes; (void)n_in; (void)out_size; (void)ws_size;
    const float* x         = (const float*)d_in[0];
    const float* qkv_w     = (const float*)d_in[1];
    const float* proj_w    = (const float*)d_in[2];
    const float* attn_gn_g = (const float*)d_in[3];
    const float* attn_gn_b = (const float*)d_in[4];
    const float* he_dw_w   = (const float*)d_in[5];
    const float* he_gn1_g  = (const float*)d_in[6];
    const float* he_gn1_b  = (const float*)d_in[7];
    const float* he_pw_w   = (const float*)d_in[8];
    const float* he_gn2_g  = (const float*)d_in[9];
    const float* he_gn2_b  = (const float*)d_in[10];
    const float* alpha     = (const float*)d_in[11];
    const float* ffn_w1    = (const float*)d_in[12];
    const float* ffn_gn1_g = (const float*)d_in[13];
    const float* ffn_gn1_b = (const float*)d_in[14];
    const float* ffn_w2    = (const float*)d_in[15];
    const float* ffn_gn2_g = (const float*)d_in[16];
    const float* ffn_gn2_b = (const float*)d_in[17];
    float* out = (float*)d_out;

    const size_t M = 1048576;           // 1M floats
    float* ws      = (float*)d_ws;
    float* ll      = ws;                // 1M
    float* ll_out  = ws + 1 * M;        // 1M
    float* high    = ws + 2 * M;        // 3M
    float* he_out  = ws + 5 * M;        // 3M
    float* bufA    = ws + 8 * M;        // 8M  (qkv, later FFN hidden)
    float* attn_o  = ws + 16 * M;       // 1M
    float* proj_r  = ws + 17 * M;       // 1M
    float* he_t1   = ws + 18 * M;       // 3M
    float* he_t2   = ws + 21 * M;       // 3M
    float* ffn_y2  = ws + 24 * M;       // 4M   (total 28M floats = 112 MB)

    dim3 blk256(256), blk128(128), blk512(512);

    // ---- DWT ----
    dwt_kernel<<<dim3((BATCH * DIM * HALFHW) / 256), blk256, 0, stream>>>(x, ll, high);

    // ---- Attention path ----
    gemm_wmma_kernel<<<dim3(HALFHW / 64, 384 / 16, BATCH), blk128, 0, stream>>>(
        qkv_w, ll, bufA, 384, DIM, HALFHW);
    l2norm_kernel<<<dim3((BATCH * 2 * HEADS * HALFHW) / 256), blk256, 0, stream>>>(bufA);
    attn_kernel<<<dim3(64, HEADS, BATCH), blk512, 0, stream>>>(bufA, attn_o);
    gemm_wmma_kernel<<<dim3(HALFHW / 64, DIM / 16, BATCH), blk128, 0, stream>>>(
        proj_w, attn_o, proj_r, DIM, DIM, HALFHW);
    gn_kernel<<<dim3(32, BATCH), blk256, 0, stream>>>(
        proj_r, ll_out, attn_gn_g, attn_gn_b, nullptr, nullptr, 0.f, DIM, HALFHW, 32, 0);

    // ---- High-frequency enhancement path ----
    dwconv_kernel<<<dim3((BATCH * 384 * HALFHW) / 256), blk256, 0, stream>>>(high, he_dw_w, he_t1);
    gn_kernel<<<dim3(32, BATCH), blk256, 0, stream>>>(
        he_t1, he_t2, he_gn1_g, he_gn1_b, nullptr, nullptr, 0.f, 384, HALFHW, 32, 1);
    gemm_wmma_kernel<<<dim3(HALFHW / 64, 384 / 16, BATCH), blk128, 0, stream>>>(
        he_pw_w, he_t2, he_t1, 384, 384, HALFHW);
    gn_kernel<<<dim3(32, BATCH), blk256, 0, stream>>>(
        he_t1, he_out, he_gn2_g, he_gn2_b, high, alpha, 0.f, 384, HALFHW, 32, 0);

    // ---- IWT + residual into out ----
    iwt_kernel<<<dim3((BATCH * DIM * HALFHW) / 256), blk256, 0, stream>>>(
        x, ll, ll_out, he_out, out);

    // ---- FFN ----
    gemm_wmma_kernel<<<dim3(FULLHW / 64, 256 / 16, BATCH), blk128, 0, stream>>>(
        ffn_w1, out, bufA, 256, DIM, FULLHW);
    gn_kernel<<<dim3(32, BATCH), blk256, 0, stream>>>(
        bufA, bufA, ffn_gn1_g, ffn_gn1_b, nullptr, nullptr, 0.f, 256, FULLHW, 32, 2);
    gemm_wmma_kernel<<<dim3(FULLHW / 64, DIM / 16, BATCH), blk128, 0, stream>>>(
        ffn_w2, bufA, ffn_y2, DIM, 256, FULLHW);
    gn_kernel<<<dim3(32, BATCH), blk256, 0, stream>>>(
        ffn_y2, out, ffn_gn2_g, ffn_gn2_b, out, nullptr, 0.1f, DIM, FULLHW, 32, 0);
}
